// MultiheadLatentAttentionTriton_89988154785798
// MI455X (gfx1250) — compile-verified, matrix operand order checked by reading, not run on hardware
//
#include <hip/hip_runtime.h>

// ---------------------------------------------------------------------------
// MLA on MI455X (gfx1250): all matmuls via v_wmma_f32_16x16x32_bf16 (wave32).
// Pipeline: cast/transpose weights -> WMMA GEMMs (Q, compress, K, V^T)
//           -> flash attention (WMMA QK^T + online softmax + WMMA PV,
//              row-max via DPP butterflies, row-sum via WMMA-with-ones)
//           -> WMMA output GEMM (fp32 out).
// GEMM tile staging uses GLOBAL_LOAD_ASYNC_TO_LDS (ASYNCcnt) when available.
// ---------------------------------------------------------------------------

typedef __attribute__((ext_vector_type(16))) __bf16 v16bf;
typedef __attribute__((ext_vector_type(8)))  __bf16 v8bf;
typedef __attribute__((ext_vector_type(8)))  float  v8f;
typedef int v4i_vs __attribute__((vector_size(16)));   // matches builtin proto

#define LDSP 40   // LDS row pitch in bf16 elems (80B: 16B-aligned, conflict-skewed)

#if defined(__gfx1250__) && __has_builtin(__builtin_amdgcn_global_load_async_to_lds_b128)
#define USE_ASYNC_LDS 1
#else
#define USE_ASYNC_LDS 0
#endif

#if USE_ASYNC_LDS
// 16-byte cache->LDS copy with no VGPR round-trip (tracked by ASYNCcnt).
// Builtin proto (from probe): (v4i __device__* src, v4i __shared__* dst,
//                              imm offset, imm cpol)
__device__ __forceinline__ void async_cp16(const __bf16* g, __bf16* l) {
    __builtin_amdgcn_global_load_async_to_lds_b128(
        (__attribute__((address_space(1))) v4i_vs*)(v4i_vs*)(void*)g,
        (__attribute__((address_space(3))) v4i_vs*)(v4i_vs*)(void*)l, 0, 0);
}
__device__ __forceinline__ void async_wait0() {
#if __has_builtin(__builtin_amdgcn_s_wait_asynccnt)
    __builtin_amdgcn_s_wait_asynccnt(0);
#else
    asm volatile("s_wait_asynccnt 0x0" ::: "memory");
#endif
}
#endif

__device__ __forceinline__ v8f wmma_bf16(v16bf a, v16bf b, v8f c) {
    // (neg_a, A, neg_b, B, c_mod, C, reuse_a, reuse_b)
    return __builtin_amdgcn_wmma_f32_16x16x32_bf16(false, a, false, b, (short)0, c,
                                                   false, false);
}

// Load one 16x32 bf16 operand tile from a row-major buffer (global or LDS).
// ISA 16-bit A/B layout: lane holds row (lane&15); K-chunks kb*8..+7 and
// 16+kb*8..+7 where kb = lane>>4. Two 16-byte vector loads per lane.
__device__ __forceinline__ v16bf ld_frag(const __bf16* base, int pitch, int row, int kb) {
    const __bf16* p = base + (size_t)row * pitch;
    v8bf lo = *(const v8bf*)(p + kb * 8);
    v8bf hi = *(const v8bf*)(p + 16 + kb * 8);
    v16bf r;
#pragma unroll
    for (int i = 0; i < 8; ++i) { r[i] = lo[i]; r[i + 8] = hi[i]; }
    return r;
}

// 16-lane max butterfly in pure VALU via DPP (no LDS bpermute traffic).
// DPP "row" = 16 lanes, so each wave32 half reduces independently — exactly
// the C-matrix layout halves.
template <int CTRL>
__device__ __forceinline__ float dpp_max_step(float v) {
    int s = __builtin_amdgcn_mov_dpp(__float_as_int(v), CTRL, 0xF, 0xF, true);
    return fmaxf(v, __int_as_float(s));
}
__device__ __forceinline__ float rmax16_dpp(float v) {
    v = dpp_max_step<0xB1>(v);   // quad_perm [1,0,3,2]  (lane ^ 1)
    v = dpp_max_step<0x4E>(v);   // quad_perm [2,3,0,1]  (lane ^ 2)
    v = dpp_max_step<0x141>(v);  // row_half_mirror      (combine quads)
    v = dpp_max_step<0x140>(v);  // row_mirror           (combine octets)
    return v;
}

// ---------------------------------------------------------------------------
// Weight cast+transpose: Wt[n][k] = bf16(W[k][n])   (W is [K,N] row-major)
// ---------------------------------------------------------------------------
__global__ __launch_bounds__(256) void tcvt_kernel(const float* __restrict__ W,
                                                   __bf16* __restrict__ Wt,
                                                   int K, int N) {
    int idx = blockIdx.x * 256 + threadIdx.x;
    if (idx >= K * N) return;
    int k = idx / N, n = idx % N;
    Wt[(size_t)n * K + k] = (__bf16)W[(size_t)k * N + n];
}

// ---------------------------------------------------------------------------
// Tiled WMMA GEMM:  out[M,N] = A[M,K] * Bt[N,K]^T + bias[N]
//   AT = float (cast to bf16 while staging) or __bf16 (async-staged)
//   OT = __bf16 or float
//   TRANS_OUT: store out[(b*N + col)*SS + s] with row = b*SS + s  (for V^T)
// Block: 256 thr = 8 waves; tile 128(M) x 64(N); K-step 32.
// ---------------------------------------------------------------------------
template <typename AT, typename OT, bool TRANS_OUT>
__global__ __launch_bounds__(256) void gemm_wmma_kernel(
    const AT* __restrict__ A, const __bf16* __restrict__ Bt,
    const float* __restrict__ bias, OT* __restrict__ out,
    int M, int N, int K, int SS) {
    __shared__ __bf16 sA[128 * LDSP];
    __shared__ __bf16 sB[64 * LDSP];

    const int tid  = threadIdx.x;
    const int wid  = tid >> 5;
    const int lane = tid & 31;
    const int m0   = blockIdx.x * 128;
    const int n0   = blockIdx.y * 64;

    const int arow = tid >> 1;          // 0..127, 2 threads per A row
    const int acol = (tid & 1) * 16;    // 0 / 16
    const int brow = tid >> 2;          // 0..63, 4 threads per Bt row
    const int bcol = (tid & 3) * 8;

    const int r  = lane & 15;
    const int kb = lane >> 4;
    const int hi = kb;

    v8f acc[4] = {};

    for (int k0 = 0; k0 < K; k0 += 32) {
        // ---- stage A tile (128 x 32) into LDS as bf16 ----
        if constexpr (sizeof(AT) == 4) {
            const float* ap = (const float*)A + (size_t)(m0 + arow) * K + k0 + acol;
            float4 f0 = *(const float4*)(ap + 0);
            float4 f1 = *(const float4*)(ap + 4);
            float4 f2 = *(const float4*)(ap + 8);
            float4 f3 = *(const float4*)(ap + 12);
            v8bf p0, p1;
            p0[0]=(__bf16)f0.x; p0[1]=(__bf16)f0.y; p0[2]=(__bf16)f0.z; p0[3]=(__bf16)f0.w;
            p0[4]=(__bf16)f1.x; p0[5]=(__bf16)f1.y; p0[6]=(__bf16)f1.z; p0[7]=(__bf16)f1.w;
            p1[0]=(__bf16)f2.x; p1[1]=(__bf16)f2.y; p1[2]=(__bf16)f2.z; p1[3]=(__bf16)f2.w;
            p1[4]=(__bf16)f3.x; p1[5]=(__bf16)f3.y; p1[6]=(__bf16)f3.z; p1[7]=(__bf16)f3.w;
            *(v8bf*)(&sA[arow * LDSP + acol])     = p0;
            *(v8bf*)(&sA[arow * LDSP + acol + 8]) = p1;
        } else {
            const __bf16* ap = (const __bf16*)A + (size_t)(m0 + arow) * K + k0 + acol;
#if USE_ASYNC_LDS
            async_cp16(ap,     &sA[arow * LDSP + acol]);
            async_cp16(ap + 8, &sA[arow * LDSP + acol + 8]);
#else
            *(v8bf*)(&sA[arow * LDSP + acol])     = *(const v8bf*)(ap);
            *(v8bf*)(&sA[arow * LDSP + acol + 8]) = *(const v8bf*)(ap + 8);
#endif
        }
        // ---- stage Bt tile (64 x 32) ----
        {
            const __bf16* bp = Bt + (size_t)(n0 + brow) * K + k0 + bcol;
#if USE_ASYNC_LDS
            async_cp16(bp, &sB[brow * LDSP + bcol]);
#else
            *(v8bf*)(&sB[brow * LDSP + bcol]) = *(const v8bf*)(bp);
#endif
        }
#if USE_ASYNC_LDS
        async_wait0();
#endif
        __syncthreads();

        v16bf af = ld_frag(&sA[(size_t)(wid * 16) * LDSP], LDSP, r, kb);
#pragma unroll
        for (int j = 0; j < 4; ++j) {
            v16bf bf = ld_frag(&sB[(size_t)(j * 16) * LDSP], LDSP, r, kb);
            acc[j] = wmma_bf16(af, bf, acc[j]);
        }
        __syncthreads();
    }

    // ---- epilogue: bias + store ----
#pragma unroll
    for (int j = 0; j < 4; ++j) {
        int colg = n0 + j * 16 + (lane & 15);
        float bv = bias[colg];
#pragma unroll
        for (int i = 0; i < 8; ++i) {
            int rowg = m0 + wid * 16 + i + 8 * hi;
            float v = acc[j][i] + bv;
            if constexpr (TRANS_OUT) {
                int bb = rowg / SS, s = rowg % SS;
                out[((size_t)bb * N + colg) * SS + s] = (OT)v;
            } else {
                out[(size_t)rowg * N + colg] = (OT)v;
            }
        }
    }
}

// ---------------------------------------------------------------------------
// Flash attention: one wave per (b, h, 32 Q rows) = two 16-row l-tiles, so
// every K/V fragment is reused by two score / PV WMMAs. Per 32-key step:
// 18 WMMA vs 16 global b128 loads.
//   Qb  : [B*L, 1024] bf16   Kb : [B*S, 1024] bf16
//   Vt  : [B, 1024, S] bf16 (V transposed)   ctx : [B*L, 1024] bf16
// ---------------------------------------------------------------------------
__global__ __launch_bounds__(256) void mla_attn_kernel(
    const __bf16* __restrict__ Qb, const __bf16* __restrict__ Kb,
    const __bf16* __restrict__ Vt, __bf16* __restrict__ ctx) {
    constexpr int L = 1024, S = 4096, LD = 1024;
    __shared__ __bf16 sP[8 * 2 * 16 * LDSP];   // per-wave two 16x32 P tiles

    const int tid  = threadIdx.x;
    const int wid  = tid >> 5;
    const int lane = tid & 31;
    const int gw   = blockIdx.x * 8 + wid;     // 0..1023
    const int lt   = gw & 31;                  // 32 l-tiles of 32 rows
    const int h    = (gw >> 5) & 15;
    const int b    = gw >> 9;

    const int r  = lane & 15;
    const int kb = lane >> 4;
    const int hi = kb;

    const __bf16* qbase = Qb + ((size_t)(b * L + lt * 32)) * LD + h * 64;
    const __bf16* kbase = Kb + ((size_t)(b * S)) * LD + h * 64;
    const __bf16* vbase = Vt + ((size_t)(b * LD + h * 64)) * S;
    __bf16* myP = &sP[wid * 2 * 16 * LDSP];

    // Resident Q fragments: 2 l-tiles x 2 d-halves
    v16bf qa[2][2];
#pragma unroll
    for (int t = 0; t < 2; ++t) {
        qa[t][0] = ld_frag(qbase + (size_t)(t * 16) * LD,      LD, r, kb);
        qa[t][1] = ld_frag(qbase + (size_t)(t * 16) * LD + 32, LD, r, kb);
    }

    // All-ones B fragment: rowsum(P) = P x ones via the matrix unit.
    v16bf ones;
#pragma unroll
    for (int i = 0; i < 16; ++i) ones[i] = (__bf16)1.0f;

    v8f acc[2][4] = {};
    float m[2][8], l[2][8];
#pragma unroll
    for (int t = 0; t < 2; ++t)
#pragma unroll
        for (int i = 0; i < 8; ++i) { m[t][i] = -3.0e38f; l[t][i] = 0.0f; }
    const float C = 0.125f * 1.44269504088896f;   // log2(e) / sqrt(D)

    for (int s0 = 0; s0 < S; s0 += 32) {
        if (s0 + 32 < S) {  // uniform branch: EXEC stays all-ones for WMMA
            __builtin_prefetch((const void*)(kbase + (size_t)(s0 + 32 + lane) * LD), 0, 0);
            __builtin_prefetch((const void*)(vbase + (size_t)(lane * 2 + 0) * S + s0 + 32), 0, 0);
            __builtin_prefetch((const void*)(vbase + (size_t)(lane * 2 + 1) * S + s0 + 32), 0, 0);
        }

        // ---- shared K fragments for this 32-key step ----
        v16bf k00 = ld_frag(kbase + (size_t)s0 * LD,             LD, r, kb);
        v16bf k01 = ld_frag(kbase + (size_t)s0 * LD + 32,        LD, r, kb);
        v16bf k10 = ld_frag(kbase + (size_t)(s0 + 16) * LD,      LD, r, kb);
        v16bf k11 = ld_frag(kbase + (size_t)(s0 + 16) * LD + 32, LD, r, kb);

        v16bf pa[2];
#pragma unroll
        for (int t = 0; t < 2; ++t) {
            // scores: 16(l) x 32(s) as two 16x16 C tiles, 4 WMMA
            v8f c0 = {}, c1 = {};
            c0 = wmma_bf16(qa[t][0], k00, c0);
            c0 = wmma_bf16(qa[t][1], k01, c0);
            c1 = wmma_bf16(qa[t][0], k10, c1);
            c1 = wmma_bf16(qa[t][1], k11, c1);

            // online softmax: row-max via DPP butterflies (pure VALU)
            float alpha[8];
#pragma unroll
            for (int i = 0; i < 8; ++i) {
                float mx = rmax16_dpp(fmaxf(c0[i], c1[i]));
                float mn = fmaxf(m[t][i], mx);
                alpha[i] = __builtin_exp2f((m[t][i] - mn) * C);
                m[t][i]  = mn;
            }
            v8f p0, p1;
#pragma unroll
            for (int i = 0; i < 8; ++i) {
                p0[i] = __builtin_exp2f((c0[i] - m[t][i]) * C);
                p1[i] = __builtin_exp2f((c1[i] - m[t][i]) * C);
            }
#pragma unroll
            for (int j = 0; j < 4; ++j)
#pragma unroll
                for (int i = 0; i < 8; ++i) acc[t][j][i] *= alpha[i];
#pragma unroll
            for (int i = 0; i < 8; ++i) l[t][i] *= alpha[i];

            // C-layout -> A-layout via wave-private LDS (DS in-order per wave)
            __bf16* tp = myP + t * 16 * LDSP;
#pragma unroll
            for (int i = 0; i < 8; ++i) {
                int row = i + 8 * hi;
                tp[row * LDSP + (lane & 15)]      = (__bf16)p0[i];
                tp[row * LDSP + 16 + (lane & 15)] = (__bf16)p1[i];
            }
            pa[t] = ld_frag(tp, LDSP, r, kb);
        }

        // row-sum via WMMA-with-ones: every lane gets its row's sum in place
#pragma unroll
        for (int t = 0; t < 2; ++t) {
            v8f z = {};
            v8f rs = wmma_bf16(pa[t], ones, z);
#pragma unroll
            for (int i = 0; i < 8; ++i) l[t][i] += rs[i];
        }

        // PV: V fragments shared across both l-tiles
#pragma unroll
        for (int j = 0; j < 4; ++j) {
            v16bf vf = ld_frag(vbase + (size_t)(j * 16) * S + s0, S, r, kb);
#pragma unroll
            for (int t = 0; t < 2; ++t) acc[t][j] = wmma_bf16(pa[t], vf, acc[t][j]);
        }
    }

    // ---- normalize + store ctx (bf16) ----
#pragma unroll
    for (int t = 0; t < 2; ++t) {
        __bf16* cbase = ctx + ((size_t)(b * L + lt * 32 + t * 16)) * LD + h * 64;
        float inv[8];
#pragma unroll
        for (int i = 0; i < 8; ++i) inv[i] = 1.0f / l[t][i];
#pragma unroll
        for (int j = 0; j < 4; ++j)
#pragma unroll
            for (int i = 0; i < 8; ++i) {
                int row = i + 8 * hi, col = j * 16 + (lane & 15);
                cbase[(size_t)row * LD + col] = (__bf16)(acc[t][j][i] * inv[i]);
            }
    }
}

// ---------------------------------------------------------------------------
// Host-side launcher
// ---------------------------------------------------------------------------
extern "C" void kernel_launch(void* const* d_in, const int* in_sizes, int n_in,
                              void* d_out, int out_size, void* d_ws, size_t ws_size,
                              hipStream_t stream) {
    (void)in_sizes; (void)n_in; (void)out_size; (void)ws_size;
    constexpr int B = 2, L = 1024, S = 4096;
    constexpr int ID = 2048, LD = 1024, RK = 128;

    const float* latent = (const float*)d_in[0];
    const float* x      = (const float*)d_in[1];
    const float* Wq     = (const float*)d_in[2];
    const float* bq     = (const float*)d_in[3];
    const float* Wc     = (const float*)d_in[4];
    const float* bc     = (const float*)d_in[5];
    const float* Wk     = (const float*)d_in[6];
    const float* bk     = (const float*)d_in[7];
    const float* Wv     = (const float*)d_in[8];
    const float* bv     = (const float*)d_in[9];
    const float* Wo     = (const float*)d_in[10];
    const float* bo     = (const float*)d_in[11];
    float* out = (float*)d_out;

    // workspace carve-up (bf16 buffers), 256B aligned
    char* ws = (char*)d_ws;
    size_t off = 0;
    auto alloc = [&](size_t elems) {
        __bf16* p = (__bf16*)(ws + off);
        off = (off + elems * sizeof(__bf16) + 255) & ~(size_t)255;
        return p;
    };
    __bf16* WqT  = alloc((size_t)LD * LD);    // [N=1024][K=1024]
    __bf16* WcT  = alloc((size_t)RK * ID);    // [128][2048]
    __bf16* WkT  = alloc((size_t)LD * RK);    // [1024][128]
    __bf16* WvT  = alloc((size_t)LD * RK);
    __bf16* WoT  = alloc((size_t)LD * LD);
    __bf16* Qb   = alloc((size_t)B * L * LD);
    __bf16* comp = alloc((size_t)B * S * RK);
    __bf16* Kb   = alloc((size_t)B * S * LD);
    __bf16* Vt   = alloc((size_t)B * LD * S);
    __bf16* ctx  = alloc((size_t)B * L * LD);

    auto cdiv = [](int a, int b) { return (a + b - 1) / b; };

    // 1) weight cast + transpose
    tcvt_kernel<<<cdiv(LD * LD, 256), 256, 0, stream>>>(Wq, WqT, LD, LD);
    tcvt_kernel<<<cdiv(ID * RK, 256), 256, 0, stream>>>(Wc, WcT, ID, RK);
    tcvt_kernel<<<cdiv(RK * LD, 256), 256, 0, stream>>>(Wk, WkT, RK, LD);
    tcvt_kernel<<<cdiv(RK * LD, 256), 256, 0, stream>>>(Wv, WvT, RK, LD);
    tcvt_kernel<<<cdiv(LD * LD, 256), 256, 0, stream>>>(Wo, WoT, LD, LD);

    // 2) Q = latent @ Wq + bq  -> bf16 [2048,1024]
    gemm_wmma_kernel<float, __bf16, false>
        <<<dim3((B * L) / 128, LD / 64), 256, 0, stream>>>(
            latent, WqT, bq, Qb, B * L, LD, LD, 0);

    // 3) compressed = x @ Wc + bc -> bf16 [8192,128]
    gemm_wmma_kernel<float, __bf16, false>
        <<<dim3((B * S) / 128, RK / 64), 256, 0, stream>>>(
            x, WcT, bc, comp, B * S, RK, ID, 0);

    // 4) K = comp @ Wk + bk -> bf16 [8192,1024]
    gemm_wmma_kernel<__bf16, __bf16, false>
        <<<dim3((B * S) / 128, LD / 64), 256, 0, stream>>>(
            comp, WkT, bk, Kb, B * S, LD, RK, 0);

    // 5) V^T = (comp @ Wv + bv)^T -> bf16 [B][1024][S]
    gemm_wmma_kernel<__bf16, __bf16, true>
        <<<dim3((B * S) / 128, LD / 64), 256, 0, stream>>>(
            comp, WvT, bv, Vt, B * S, LD, RK, S);

    // 6) flash attention -> ctx bf16 [2048,1024] (1024 waves, 8 per block)
    mla_attn_kernel<<<(B * 16 * (L / 32)) / 8, 256, 0, stream>>>(Qb, Kb, Vt, ctx);

    // 7) out = ctx @ Wo + bo -> fp32
    gemm_wmma_kernel<__bf16, float, false>
        <<<dim3((B * L) / 128, LD / 64), 256, 0, stream>>>(
            ctx, WoT, bo, out, B * L, LD, LD, 0);
}
